// MRCGNN_78572131713265
// MI455X (gfx1250) — compile-verified, matrix operand order checked by reading, not run on hardware
//
#include <hip/hip_runtime.h>
#include <hip/hip_bf16.h>

#define N_NODES 10000
#define F_INx   128
#define H1x     64
#define H2x     32
#define NREL    65
#define E_Ox    320000
#define E_Sx    160000
#define NBx     4
#define BPAIR   4096
#define RCHUNK  4   // relations per aggregation chunk (keeps workspace < 48 MB)

typedef __attribute__((ext_vector_type(16))) __bf16 v16bf;
typedef __attribute__((ext_vector_type(8)))  float  v8f;

__device__ __forceinline__ __bf16 f2bf(float f) {
    // round-to-nearest-even f32 -> bf16
    unsigned u = __float_as_uint(f);
    u += 0x7FFFu + ((u >> 16) & 1u);
    unsigned short h = (unsigned short)(u >> 16);
    return __builtin_bit_cast(__bf16, h);
}

// ---------------- utility kernels ----------------

__global__ void zero_f_kernel(float* __restrict__ p, long n) {
    long i = (long)blockIdx.x * blockDim.x + threadIdx.x;
    if (i < n) p[i] = 0.0f;
}

__global__ void deg_kernel(const int* __restrict__ dst, const int* __restrict__ et,
                           int E, float* __restrict__ deg) {
    int i = blockIdx.x * blockDim.x + threadIdx.x;
    if (i < E) atomicAdd(&deg[(long)dst[i] * NREL + et[i]], 1.0f);
}

__global__ void inv_kernel(float* __restrict__ p, long n) {
    long i = (long)blockIdx.x * blockDim.x + threadIdx.x;
    if (i < n) p[i] = 1.0f / fmaxf(p[i], 1.0f);
}

// scatter x[src]*norm into agg[rel_local][dst][0:Fin]; blockDim.x == Fin
__global__ void scatter_kernel(const int* __restrict__ src, const int* __restrict__ dst,
                               const int* __restrict__ et, int E,
                               const float* __restrict__ X, int Fin,
                               const float* __restrict__ norm,
                               int r0, int rc, float* __restrict__ agg) {
    int e = blockIdx.x;
    if (e >= E) return;
    int r = et[e] - r0;
    if (r < 0 || r >= rc) return;
    int s = src[e], d = dst[e];
    float nv = norm[(long)d * NREL + (r + r0)];
    int f = threadIdx.x;
    atomicAdd(&agg[((size_t)r * N_NODES + d) * Fin + f], X[(size_t)s * Fin + f] * nv);
}

// feats[b] = [embeds[d1](160), features1[d1](128), embeds[d2](160), features1[d2](128)]
__global__ void gather_feats_kernel(const int* __restrict__ d1, const int* __restrict__ d2,
                                    const float* __restrict__ embeds,
                                    const float* __restrict__ feat1,
                                    float* __restrict__ feats) {
    int b = blockIdx.x;
    int t = threadIdx.x;  // 576 threads
    int n1 = d1[b], n2 = d2[b];
    float v;
    if (t < 160)       v = embeds[(size_t)n1 * 160 + t];
    else if (t < 288)  v = feat1 [(size_t)n1 * 128 + (t - 160)];
    else if (t < 448)  v = embeds[(size_t)n2 * 160 + (t - 288)];
    else               v = feat1 [(size_t)n2 * 128 + (t - 448)];
    feats[(size_t)b * 576 + t] = v;
}

// ---------------- B pre-pack into WMMA fragment order ----------------
// bpack[tn][kblk][lane][0..15] = the exact v16bf B operand for v_wmma_f32_16x16x32_bf16.
// Packed for tNpad = 4*ceil(tN/4) tiles; OOB columns are clamped duplicates that are
// loaded (safely) but never written back.
__global__ void pack_b_kernel(const float* __restrict__ Bm, int ldb, int Nc, int nK32,
                              __bf16* __restrict__ bp) {
    int kblk = blockIdx.x;
    int tn   = blockIdx.y;
    int lane = threadIdx.x;
    int half = lane >> 4;
    int l    = lane & 15;
    int n   = tn * 16 + l;
    int ncl = n < Nc ? n : Nc - 1;
    const float* src = Bm + (size_t)(kblk * 32 + half * 8) * ldb + ncl;
    __bf16* dst = bp + (((size_t)tn * nK32 + kblk) * 32 + lane) * 16;
    #pragma unroll
    for (int i = 0; i < 8; ++i) {
        dst[i]     = f2bf(src[(size_t)i * ldb]);
        dst[8 + i] = f2bf(src[(size_t)(16 + i) * ldb]);
    }
}

// ---------------- bf16 WMMA GEMM: one wave -> 16x64 of C (4 n-tiles) ----------------
// C[M,Nc] (+)= A * B.  A blocked-K layout:
//   A(m, kb*kBlock + kin) = A[kb*aBlockStride + m*aRowStride + kin],  K = nKBlocks*kBlock
// B pre-packed in fragment order. One A fragment feeds 4 WMMAs (4 accumulators),
// amortizing A loads and f32->bf16 conversion 4x.
// Grid: (ceil(tM/4), ceil(tN/4)); 128 threads = 4 waves over 4 consecutive row tiles.
__device__ __forceinline__ void epilogue_tile(float* __restrict__ C, int ldc,
                                              const float* __restrict__ bias,
                                              int M, int Nc, int tm, int half, int col,
                                              v8f acc, int accumulate, int act) {
    if (col < Nc) {
        float bv = bias ? bias[col] : 0.0f;
        #pragma unroll
        for (int j = 0; j < 8; ++j) {
            int row = tm * 16 + half * 8 + j;
            if (row < M) {
                size_t ci = (size_t)row * ldc + col;
                float v = acc[j] + bv;
                if (accumulate) v += C[ci];
                if (act == 1)      v = fmaxf(v, 0.0f);
                else if (act == 2) v = v > 0.0f ? v : (__expf(v) - 1.0f);
                C[ci] = v;
            }
        }
    }
}

__global__ void wmma_gemm_kernel(const float* __restrict__ A, int aRowStride, int kBlock,
                                 long aBlockStride, int nKBlocks, int nK32,
                                 const __bf16* __restrict__ Bpk,
                                 float* __restrict__ C, int ldc,
                                 const float* __restrict__ bias,
                                 int M, int Nc, int tM,
                                 int accumulate, int act) {
    int wave = threadIdx.x >> 5;
    int lane = threadIdx.x & 31;
    int tm = blockIdx.x * 4 + wave;          // row tile per wave
    if (tm >= tM) return;
    int grp  = blockIdx.y;                   // group of 4 consecutive n-tiles
    int half = lane >> 4;
    int l    = lane & 15;

    int m = tm * 16 + l;
    if (m >= M) m = M - 1;                   // clamp (M multiple of 16 at all call sites)

    const float* Ab = A + (size_t)m * aRowStride + half * 8;
    size_t tileStride = (size_t)nK32 * 512;  // bpack elements per n-tile
    const __bf16* bp0 = Bpk + (size_t)(grp * 4) * tileStride + (size_t)lane * 16;
    const __bf16* bp1 = bp0 + tileStride;
    const __bf16* bp2 = bp1 + tileStride;
    const __bf16* bp3 = bp2 + tileStride;

    v8f acc0 = {}, acc1 = {}, acc2 = {}, acc3 = {};
    for (int kb = 0; kb < nKBlocks; ++kb, Ab += aBlockStride) {
        const float* Ap = Ab;
        #pragma unroll 2
        for (int kin = 0; kin < kBlock; kin += 32, Ap += 32,
                 bp0 += 512, bp1 += 512, bp2 += 512, bp3 += 512) {
            float4 p0 = *(const float4*)(Ap);
            float4 p1 = *(const float4*)(Ap + 4);
            float4 q0 = *(const float4*)(Ap + 16);
            float4 q1 = *(const float4*)(Ap + 20);
            v16bf a;
            a[0]  = f2bf(p0.x); a[1]  = f2bf(p0.y); a[2]  = f2bf(p0.z); a[3]  = f2bf(p0.w);
            a[4]  = f2bf(p1.x); a[5]  = f2bf(p1.y); a[6]  = f2bf(p1.z); a[7]  = f2bf(p1.w);
            a[8]  = f2bf(q0.x); a[9]  = f2bf(q0.y); a[10] = f2bf(q0.z); a[11] = f2bf(q0.w);
            a[12] = f2bf(q1.x); a[13] = f2bf(q1.y); a[14] = f2bf(q1.z); a[15] = f2bf(q1.w);
            v16bf b0 = *(const v16bf*)bp0;
            v16bf b1 = *(const v16bf*)bp1;
            v16bf b2 = *(const v16bf*)bp2;
            v16bf b3 = *(const v16bf*)bp3;
            acc0 = __builtin_amdgcn_wmma_f32_16x16x32_bf16(false, a, false, b0, (short)0, acc0, false, false);
            acc1 = __builtin_amdgcn_wmma_f32_16x16x32_bf16(false, a, false, b1, (short)0, acc1, false, false);
            acc2 = __builtin_amdgcn_wmma_f32_16x16x32_bf16(false, a, false, b2, (short)0, acc2, false, false);
            acc3 = __builtin_amdgcn_wmma_f32_16x16x32_bf16(false, a, false, b3, (short)0, acc3, false, false);
        }
    }

    int nBase = grp * 64 + l;
    epilogue_tile(C, ldc, bias, M, Nc, tm, half, nBase,      acc0, accumulate, act);
    epilogue_tile(C, ldc, bias, M, Nc, tm, half, nBase + 16, acc1, accumulate, act);
    epilogue_tile(C, ldc, bias, M, Nc, tm, half, nBase + 32, acc2, accumulate, act);
    epilogue_tile(C, ldc, bias, M, Nc, tm, half, nBase + 48, acc3, accumulate, act);
}

// ---------------- host-side helpers ----------------

static inline void zero_buf(float* p, long n, hipStream_t s) {
    zero_f_kernel<<<(unsigned)((n + 255) / 256), 256, 0, s>>>(p, n);
}

static inline void launch_gemm(const float* A, int aRowStride, int kBlock, long aBlockStride,
                               int nKBlocks,
                               const float* Bm, int ldb, __bf16* bpack,
                               float* C, int ldc,
                               const float* bias, int M, int Nc,
                               int accumulate, int act, hipStream_t s) {
    int tM = (M + 15) / 16, tN = (Nc + 15) / 16;
    int nGroups = (tN + 3) / 4;
    int tNpad = nGroups * 4;
    int nK32 = (nKBlocks * kBlock) / 32;
    pack_b_kernel<<<dim3((unsigned)nK32, (unsigned)tNpad), 32, 0, s>>>(Bm, ldb, Nc, nK32, bpack);
    dim3 grid((unsigned)((tM + 3) / 4), (unsigned)nGroups);
    wmma_gemm_kernel<<<grid, 128, 0, s>>>(A, aRowStride, kBlock, aBlockStride, nKBlocks, nK32,
                                          bpack, C, ldc, bias, M, Nc, tM, accumulate, act);
}

// One RGCN layer: C[N, Fout] (ldc) = act( sum_r mean-agg_r @ W_r + X @ Wroot + bias )
static void rgcn_layer(const float* X, int Fin, int Fout,
                       const float* Wr, const float* Wroot, const float* bias,
                       const int* src, const int* dst, const int* et, int E,
                       const float* norm, float* agg, __bf16* bpack,
                       float* C, int ldc, int act, hipStream_t s) {
    int nch = (NREL + RCHUNK - 1) / RCHUNK;
    for (int c = 0; c < nch; ++c) {
        int r0 = c * RCHUNK;
        int rc = NREL - r0 < RCHUNK ? NREL - r0 : RCHUNK;
        zero_buf(agg, (long)rc * N_NODES * Fin, s);
        scatter_kernel<<<(unsigned)E, Fin, 0, s>>>(src, dst, et, E, X, Fin, norm, r0, rc, agg);
        // C (+)= [agg_r0 | ... | agg_r0+rc-1] @ W[r0..r0+rc-1]
        launch_gemm(agg, Fin, Fin, (long)N_NODES * Fin, rc,
                    Wr + (size_t)r0 * Fin * Fout, Fout, bpack,
                    C, ldc, nullptr, N_NODES, Fout,
                    /*accumulate=*/(c > 0), /*act=*/0, s);
    }
    // root transform + bias + activation (last accumulation into C)
    launch_gemm(X, Fin, Fin, 0, 1, Wroot, Fout, bpack, C, ldc, bias,
                N_NODES, Fout, /*accumulate=*/1, act, s);
}

extern "C" void kernel_launch(void* const* d_in, const int* in_sizes, int n_in,
                              void* d_out, int out_size, void* d_ws, size_t ws_size,
                              hipStream_t stream) {
    (void)in_sizes; (void)n_in; (void)out_size; (void)ws_size;

    const float* x_o     = (const float*)d_in[0];
    const int*   ei_o    = (const int*)  d_in[1];
    const int*   et_o    = (const int*)  d_in[2];
    const float* x_s     = (const float*)d_in[3];
    const int*   ei_s    = (const int*)  d_in[4];
    const int*   et_s    = (const int*)  d_in[5];
    const int*   idx     = (const int*)  d_in[6];
    const float* feat1   = (const float*)d_in[7];
    const float* W1      = (const float*)d_in[8];
    const float* root1   = (const float*)d_in[9];
    const float* b1      = (const float*)d_in[10];
    const float* W2      = (const float*)d_in[11];
    const float* root2   = (const float*)d_in[12];
    const float* b2      = (const float*)d_in[13];
    const float* Ws1     = (const float*)d_in[14];
    const float* roots1  = (const float*)d_in[15];
    const float* bs1     = (const float*)d_in[16];
    const float* Ws2     = (const float*)d_in[17];
    const float* roots2  = (const float*)d_in[18];
    const float* bs2     = (const float*)d_in[19];
    const float* mlp_w1  = (const float*)d_in[20];
    const float* mlp_b1  = (const float*)d_in[21];
    const float* mlp_w2  = (const float*)d_in[22];
    const float* mlp_b2  = (const float*)d_in[23];
    const float* mlp_w3  = (const float*)d_in[24];
    const float* mlp_b3  = (const float*)d_in[25];
    float* out = (float*)d_out;

    // workspace layout (all offsets multiples of 4 floats -> 16B aligned)
    float* ws     = (float*)d_ws;
    float* norm   = ws;                                        // N*R       = 650,000
    float* agg    = norm   + (size_t)N_NODES * NREL;           // RC*N*F_IN = 5,120,000
    float* h1b    = agg    + (size_t)RCHUNK * N_NODES * F_INx; // N*H1      =   640,000
    float* embeds = h1b    + (size_t)N_NODES * H1x;            // N*160     = 1,600,000
    float* feats  = embeds + (size_t)N_NODES * 5 * H2x;        // B*576     = 2,359,296
    float* mh1    = feats  + (size_t)BPAIR * 576;              // B*256
    float* mh2    = mh1    + (size_t)BPAIR * 256;              // B*128
    __bf16* bpack = (__bf16*)(mh2 + (size_t)BPAIR * 128);      // <= 147,456 bf16 (288 KB)

    for (int g = 0; g < 1 + NBx; ++g) {
        const int *src, *dst, *et; int E;
        const float *X, *W1g, *r1g, *b1g, *W2g, *r2g, *b2g;
        if (g == 0) {
            src = ei_o; dst = ei_o + E_Ox; et = et_o; E = E_Ox;
            X = x_o; W1g = W1; r1g = root1; b1g = b1; W2g = W2; r2g = root2; b2g = b2;
        } else {
            int b = g - 1;
            src = ei_s + (size_t)b * 2 * E_Sx; dst = src + E_Sx;
            et  = et_s + (size_t)b * E_Sx;     E = E_Sx;
            X   = x_s  + (size_t)b * N_NODES * F_INx;
            W1g = Ws1    + (size_t)b * NREL * F_INx * H1x;
            r1g = roots1 + (size_t)b * F_INx * H1x;
            b1g = bs1    + (size_t)b * H1x;
            W2g = Ws2    + (size_t)b * NREL * H1x * H2x;
            r2g = roots2 + (size_t)b * H1x * H2x;
            b2g = bs2    + (size_t)b * H2x;
        }
        // per-(dst, rel) mean normalization: norm = 1/max(deg,1)
        zero_buf(norm, (long)N_NODES * NREL, stream);
        deg_kernel<<<(unsigned)((E + 255) / 256), 256, 0, stream>>>(dst, et, E, norm);
        inv_kernel<<<(unsigned)(((long)N_NODES * NREL + 255) / 256), 256, 0, stream>>>(
            norm, (long)N_NODES * NREL);

        // layer 1: [N,128] -> relu -> [N,64]
        rgcn_layer(X, F_INx, H1x, W1g, r1g, b1g, src, dst, et, E, norm, agg, bpack,
                   h1b, H1x, /*act=relu*/1, stream);
        // layer 2: [N,64] -> [N,32], written into its embeds column slice
        rgcn_layer(h1b, H1x, H2x, W2g, r2g, b2g, src, dst, et, E, norm, agg, bpack,
                   embeds + g * H2x, 5 * H2x, /*act=none*/0, stream);
    }

    // pair feature gather: [B,576]
    gather_feats_kernel<<<BPAIR, 576, 0, stream>>>(idx, idx + BPAIR, embeds, feat1, feats);

    // MLP head: 576 -> 256 (elu) -> 128 (elu) -> 65
    launch_gemm(feats, 576, 576, 0, 1, mlp_w1, 256, bpack, mh1, 256, mlp_b1,
                BPAIR, 256, 0, 2, stream);
    launch_gemm(mh1,   256, 256, 0, 1, mlp_w2, 128, bpack, mh2, 128, mlp_b2,
                BPAIR, 128, 0, 2, stream);
    launch_gemm(mh2,   128, 128, 0, 1, mlp_w3,  65, bpack, out,  65, mlp_b3,
                BPAIR,  65, 0, 0, stream);
}
